// Qwen3MoeSparseMoeBlockEP_58858231824407
// MI455X (gfx1250) — compile-verified, matrix operand order checked by reading, not run on hardware
//
#include <hip/hip_runtime.h>
#include <hip/hip_bf16.h>

// ---------------------------------------------------------------------------
// Qwen3 MoE sparse block (top-2 of 8 experts), MI455X / gfx1250, wave32.
// GEMMs: v_wmma_f32_16x16x32_bf16. A-staging: async-to-LDS (mlp1, gathered
// rows) and Tensor Data Mover (mlp2, contiguous rows with HW row padding).
// ---------------------------------------------------------------------------

#define T_TOK  2048
#define HID    2048
#define INTERF 768
#define NEXP   8

typedef __attribute__((ext_vector_type(16))) __bf16 v16bf;
typedef __attribute__((ext_vector_type(8)))  __bf16 v8bf;
typedef __attribute__((ext_vector_type(8)))  float  v8f;
typedef __attribute__((ext_vector_type(4)))  unsigned int u32x4;
typedef __attribute__((ext_vector_type(8)))  int i32x8;
typedef __attribute__((ext_vector_type(4)))  int i32x4;

#define BM   64   // block tile rows (tokens/slots)
#define BN   64   // block tile cols
#define BK   32   // K step (matches 16x16x32 wmma)
#define LDST 40   // LDS row stride in bf16 elems (80B: 16B aligned, conflict-free)

// Assemble a 16-element bf16 fragment from two 16B LDS chunks.
// Per-lane layout (A and B identically): elems 0..7 -> K = 8h..8h+7,
// elems 8..15 -> K = 16+8h..23+8h, where h = lane>>4.
__device__ __forceinline__ v16bf load_frag(const __bf16* p) {
  v8bf lo = *(const v8bf*)p;
  v8bf hi = *(const v8bf*)(p + 16);
  return __builtin_shufflevector(lo, hi, 0, 1, 2, 3, 4, 5, 6, 7,
                                         8, 9, 10, 11, 12, 13, 14, 15);
}

__device__ __forceinline__ v8f wmma_bf16(v16bf a, v16bf b, v8f c) {
  return __builtin_amdgcn_wmma_f32_16x16x32_bf16(false, a, false, b,
                                                 (short)0, c, false, false);
}

// Async 16B global->LDS copy (GVS mode), tracked by ASYNCcnt.
__device__ __forceinline__ void async_copy_b128(unsigned lds_addr,
                                                unsigned voff,
                                                unsigned long long sbase) {
  asm volatile("global_load_async_to_lds_b128 %0, %1, %2"
               :: "v"(lds_addr), "v"(voff), "s"(sbase) : "memory");
}
__device__ __forceinline__ void wait_asynccnt0() {
  asm volatile("s_wait_asynccnt 0x0" ::: "memory");
}

// TDM: 2D tile (tileK x tileRows) of a bf16 row-major tensor -> LDS, with
// hardware row padding 64B -> 80B (pad_interval=16 DW, pad_amount=4 DW).
__device__ __forceinline__ void tdm_load_tile_bf16(
    unsigned lds_addr, unsigned long long gaddr, int rowsAvail,
    int tensorRowElems, int tileK, int tileRows) {
  u32x4 g0;
  g0[0] = 1u;                                   // count=1 (valid descriptor)
  g0[1] = lds_addr;                             // lds_addr (bytes)
  g0[2] = (unsigned)gaddr;                      // global_addr[31:0]
  g0[3] = (unsigned)(gaddr >> 32) | (2u << 30); // global_addr[56:32] | type=2
  i32x8 g1;
  g1[0] = (1 << 16)        // data_size = 2 bytes (bf16)
        | (1 << 20)        // pad_enable
        | (3 << 22)        // pad_interval: 16 DWORDs (= 64B tile row)
        | (3 << 25);       // pad_amount: 4 DWORDs (= 16B -> 80B row stride)
  g1[1] = (tensorRowElems & 0xFFFF) << 16;           // tensor_dim0 lo16
  g1[2] = (tensorRowElems >> 16) & 0xFFFF;           // tensor_dim0 hi16
  g1[2] |= (rowsAvail & 0xFFFF) << 16;               // tensor_dim1 lo16
  g1[3] = ((rowsAvail >> 16) & 0xFFFF) | (tileK << 16);  // dim1 hi | tile_dim0
  g1[4] = tileRows;                                  // tile_dim1 (tile_dim2=0)
  g1[5] = tensorRowElems;                            // tensor_dim0_stride lo32
  g1[6] = 0;                                         // stride hi / dim1_stride
  g1[7] = 0;
  i32x4 gz = {0, 0, 0, 0};
#if __clang_major__ >= 23
  i32x8 gz8 = {0, 0, 0, 0, 0, 0, 0, 0};
  __builtin_amdgcn_tensor_load_to_lds(g0, g1, gz, gz, gz8, 0);
#else
  __builtin_amdgcn_tensor_load_to_lds(g0, g1, gz, gz, 0);
#endif
}

// ---------------------------------------------------------------------------
// 0) zero per-expert counters
// ---------------------------------------------------------------------------
__global__ void moe_init_kernel(int* cnt) {
  if (threadIdx.x < NEXP) cnt[threadIdx.x] = 0;
}

// ---------------------------------------------------------------------------
// 1) router: logits, softmax(fp32), top-2, renormalize. One wave per token.
// ---------------------------------------------------------------------------
__global__ __launch_bounds__(256) void moe_router_kernel(
    const float* __restrict__ x, const float* __restrict__ gate_w,
    float* __restrict__ logits_out, int* __restrict__ tokExp,
    float* __restrict__ tokW, int* __restrict__ cnt) {
  const int lane = threadIdx.x & 31;
  const int wave = threadIdx.x >> 5;
  const int t = blockIdx.x * 8 + wave;

  float acc[NEXP];
#pragma unroll
  for (int e = 0; e < NEXP; ++e) acc[e] = 0.0f;

  const float* xr = x + (size_t)t * HID;
  for (int h = lane * 4; h < HID; h += 32 * 4) {
    const float4 xv = *(const float4*)(xr + h);
#pragma unroll
    for (int e = 0; e < NEXP; ++e) {
      const float4 g = *(const float4*)(gate_w + (size_t)e * HID + h);
      acc[e] += xv.x * g.x + xv.y * g.y + xv.z * g.z + xv.w * g.w;
    }
  }
#pragma unroll
  for (int e = 0; e < NEXP; ++e)
    for (int off = 16; off > 0; off >>= 1)
      acc[e] += __shfl_xor(acc[e], off, 32);

  if (lane < NEXP) logits_out[(size_t)t * NEXP + lane] = acc[lane];

  if (lane == 0) {
    float mx = acc[0];
#pragma unroll
    for (int e = 1; e < NEXP; ++e) mx = fmaxf(mx, acc[e]);
    float p[NEXP];
    float s = 0.0f;
#pragma unroll
    for (int e = 0; e < NEXP; ++e) { p[e] = expf(acc[e] - mx); s += p[e]; }
    const float inv_s = 1.0f / s;
#pragma unroll
    for (int e = 0; e < NEXP; ++e) p[e] *= inv_s;

    int e0 = 0; float b0 = p[0];
#pragma unroll
    for (int e = 1; e < NEXP; ++e) if (p[e] > b0) { b0 = p[e]; e0 = e; }
    int e1 = (e0 == 0) ? 1 : 0; float b1 = p[e1];
#pragma unroll
    for (int e = 0; e < NEXP; ++e)
      if (e != e0 && p[e] > b1) { b1 = p[e]; e1 = e; }

    const float inv = 1.0f / (b0 + b1);
    tokExp[t * 2 + 0] = e0; tokW[t * 2 + 0] = b0 * inv;
    tokExp[t * 2 + 1] = e1; tokW[t * 2 + 1] = b1 * inv;
    atomicAdd(cnt + e0, 1);
    atomicAdd(cnt + e1, 1);
  }
}

// ---------------------------------------------------------------------------
// 2) exclusive scan of 8 counts + reset cursors
// ---------------------------------------------------------------------------
__global__ void moe_scan_kernel(const int* __restrict__ cnt,
                                int* __restrict__ base,
                                int* __restrict__ cursor) {
  if (threadIdx.x == 0) {
    int run = 0;
    for (int e = 0; e < NEXP; ++e) { base[e] = run; run += cnt[e]; cursor[e] = 0; }
  }
}

// ---------------------------------------------------------------------------
// 3) build compacted expert-major slot lists
// ---------------------------------------------------------------------------
__global__ void moe_build_kernel(const int* __restrict__ tokExp,
                                 const float* __restrict__ tokW,
                                 const int* __restrict__ base,
                                 int* __restrict__ cursor,
                                 int* __restrict__ slotToken,
                                 float* __restrict__ slotWeight,
                                 int* __restrict__ tok2slot) {
  const int t = blockIdx.x * blockDim.x + threadIdx.x;
  if (t >= T_TOK) return;
#pragma unroll
  for (int k = 0; k < 2; ++k) {
    const int e = tokExp[t * 2 + k];
    const int pos = atomicAdd(cursor + e, 1);
    const int slot = base[e] + pos;
    slotToken[slot] = t;
    slotWeight[slot] = tokW[t * 2 + k];
    tok2slot[t * 2 + k] = slot;
  }
}

// ---------------------------------------------------------------------------
// 4) cast activations fp32 -> bf16
// ---------------------------------------------------------------------------
__global__ __launch_bounds__(256) void moe_castx_kernel(
    const float* __restrict__ x, __bf16* __restrict__ xbf) {
  const size_t i = ((size_t)blockIdx.x * 256 + threadIdx.x) * 8;
  const float4 a = *(const float4*)(x + i);
  const float4 b = *(const float4*)(x + i + 4);
  v8bf o;
  o[0] = (__bf16)a.x; o[1] = (__bf16)a.y; o[2] = (__bf16)a.z; o[3] = (__bf16)a.w;
  o[4] = (__bf16)b.x; o[5] = (__bf16)b.y; o[6] = (__bf16)b.z; o[7] = (__bf16)b.w;
  *(v8bf*)(xbf + i) = o;
}

// ---------------------------------------------------------------------------
// 5) fused gate/up GEMM + SwiGLU. A-tile: global_load_async_to_lds_b128
//    (per-lane gather of token rows, overlaps with fp32 B staging + cvt).
// ---------------------------------------------------------------------------
__global__ __launch_bounds__(256) void moe_mlp1_kernel(
    const __bf16* __restrict__ xbf, const float* __restrict__ w_gate,
    const float* __restrict__ w_up, const int* __restrict__ slotToken,
    const int* __restrict__ baseArr, const int* __restrict__ cntArr,
    __bf16* __restrict__ hbuf) {
  const int e = blockIdx.z;
  const int cnt = cntArr[e];
  const int m0 = blockIdx.y * BM;
  if (m0 >= cnt) return;  // uniform per block: EXEC stays all-ones for WMMA
  const int base = baseArr[e];
  const int n0 = blockIdx.x * BN;

  __shared__ __bf16 ldsA[BM * LDST];
  __shared__ __bf16 ldsBg[BN * LDST];
  __shared__ __bf16 ldsBu[BN * LDST];

  const int tid = threadIdx.x;
  const int lane = tid & 31;
  const int wave = tid >> 5;
  const int wm = wave & 3;   // 16-row slab within 64
  const int wn = wave >> 2;  // 32-col slab within 64
  const int half = lane >> 4;
  const int ml = lane & 15;

  // A staging: thread -> (row, 16B chunk); async copy of gathered token rows
  const int rowA = tid >> 2;
  const int chunkA = tid & 3;
  int srcRow = m0 + rowA; if (srcRow >= cnt) srcRow = cnt - 1;
  const int tok = slotToken[base + srcRow];
  const unsigned ldsA_dst =
      (unsigned)(uintptr_t)&ldsA[rowA * LDST + chunkA * 8];
  const unsigned long long xbase = (unsigned long long)(uintptr_t)xbf;
  const unsigned voffBase =
      (unsigned)(((size_t)tok * HID + (size_t)chunkA * 8) * 2);

  // B staging: thread -> col f (coalesced), 8 K rows; store transposed [f][k]
  const int fB = tid & 63;
  const int kB0 = (tid >> 6) * 8;
  const float* gSrc = w_gate + (size_t)e * HID * INTERF + n0 + fB;
  const float* uSrc = w_up   + (size_t)e * HID * INTERF + n0 + fB;

  v8f cg0 = {}, cg1 = {}, cu0 = {}, cu1 = {};
  const int mfrag = wm * 16;
  const int nfrag = wn * 32;

  for (int k0 = 0; k0 < HID; k0 += BK) {
    async_copy_b128(ldsA_dst, voffBase + (unsigned)(k0 * 2), xbase);
#pragma unroll
    for (int i = 0; i < 8; ++i) {
      const int k = kB0 + i;
      ldsBg[fB * LDST + k] = (__bf16)gSrc[(size_t)(k0 + k) * INTERF];
      ldsBu[fB * LDST + k] = (__bf16)uSrc[(size_t)(k0 + k) * INTERF];
    }
    if (k0 + BK < HID) {  // pull next weight K-slab toward L2 early
      __builtin_prefetch(gSrc + (size_t)(k0 + BK + kB0) * INTERF, 0, 1);
      __builtin_prefetch(uSrc + (size_t)(k0 + BK + kB0) * INTERF, 0, 1);
    }
    wait_asynccnt0();
    __syncthreads();

    const v16bf a   = load_frag(&ldsA[(mfrag + ml) * LDST + half * 8]);
    const v16bf bg0 = load_frag(&ldsBg[(nfrag + ml) * LDST + half * 8]);
    const v16bf bg1 = load_frag(&ldsBg[(nfrag + 16 + ml) * LDST + half * 8]);
    const v16bf bu0 = load_frag(&ldsBu[(nfrag + ml) * LDST + half * 8]);
    const v16bf bu1 = load_frag(&ldsBu[(nfrag + 16 + ml) * LDST + half * 8]);
    cg0 = wmma_bf16(a, bg0, cg0);
    cg1 = wmma_bf16(a, bg1, cg1);
    cu0 = wmma_bf16(a, bu0, cu0);
    cu1 = wmma_bf16(a, bu1, cu1);
    __syncthreads();
  }

  // SwiGLU epilogue: h = silu(g) * u, bf16 into compacted hbuf[slot][INTER]
#pragma unroll
  for (int r = 0; r < 8; ++r) {
    const int mloc = mfrag + r + 8 * half;  // C: VGPR r -> M=r (+8 hi lanes)
    const int row = m0 + mloc;
    if (row < cnt) {
      const size_t slot = (size_t)(base + row);
      const float g0 = cg0[r];
      const float h0 = g0 * cu0[r] * (1.0f / (1.0f + __expf(-g0)));
      hbuf[slot * INTERF + (n0 + nfrag + ml)] = (__bf16)h0;
      const float g1 = cg1[r];
      const float h1 = g1 * cu1[r] * (1.0f / (1.0f + __expf(-g1)));
      hbuf[slot * INTERF + (n0 + nfrag + 16 + ml)] = (__bf16)h1;
    }
  }
}

// ---------------------------------------------------------------------------
// 6) down projection GEMM. A-tile via TDM tensor_load_to_lds: contiguous
//    slot rows of hbuf, HW pad 64B->80B rows, zero-fill past cnt.
// ---------------------------------------------------------------------------
__global__ __launch_bounds__(256) void moe_mlp2_kernel(
    const __bf16* __restrict__ hbuf, const float* __restrict__ w_down,
    const int* __restrict__ baseArr, const int* __restrict__ cntArr,
    const float* __restrict__ slotWeight, float* __restrict__ ybuf) {
  const int e = blockIdx.z;
  const int cnt = cntArr[e];
  const int m0 = blockIdx.y * BM;
  if (m0 >= cnt) return;
  const int base = baseArr[e];
  const int n0 = blockIdx.x * BN;

  __shared__ __bf16 ldsA[BM * LDST];
  __shared__ __bf16 ldsB[BN * LDST];

  const int tid = threadIdx.x;
  const int lane = tid & 31;
  const int wave = tid >> 5;
  const int wm = wave & 3;
  const int wn = wave >> 2;
  const int half = lane >> 4;
  const int ml = lane & 15;

  const int rowsAvail = cnt - m0;  // TDM zero-fills rows beyond this
  const unsigned ldsA_base = (unsigned)(uintptr_t)&ldsA[0];
  const unsigned long long gaBase =
      (unsigned long long)(uintptr_t)(hbuf + (size_t)(base + m0) * INTERF);

  const int fB = tid & 63;
  const int kB0 = (tid >> 6) * 8;
  const float* dSrc = w_down + (size_t)e * INTERF * HID + n0 + fB;

  v8f c0 = {}, c1 = {};
  const int mfrag = wm * 16;
  const int nfrag = wn * 32;

  for (int k0 = 0; k0 < INTERF; k0 += BK) {
    if (wave == 0)  // one TDM descriptor per block per K-step (EXEC ignored)
      tdm_load_tile_bf16(ldsA_base, gaBase + (unsigned long long)(k0 * 2),
                         rowsAvail, INTERF, BK, BM);
#pragma unroll
    for (int i = 0; i < 8; ++i) {
      const int k = kB0 + i;
      ldsB[fB * LDST + k] = (__bf16)dSrc[(size_t)(k0 + k) * HID];
    }
    if (k0 + BK < INTERF)
      __builtin_prefetch(dSrc + (size_t)(k0 + BK + kB0) * HID, 0, 1);
    __builtin_amdgcn_s_wait_tensorcnt((short)0);
    __syncthreads();

    const v16bf a  = load_frag(&ldsA[(mfrag + ml) * LDST + half * 8]);
    const v16bf b0 = load_frag(&ldsB[(nfrag + ml) * LDST + half * 8]);
    const v16bf b1 = load_frag(&ldsB[(nfrag + 16 + ml) * LDST + half * 8]);
    c0 = wmma_bf16(a, b0, c0);
    c1 = wmma_bf16(a, b1, c1);
    __syncthreads();
  }

#pragma unroll
  for (int r = 0; r < 8; ++r) {
    const int mloc = mfrag + r + 8 * half;
    const int row = m0 + mloc;
    if (row < cnt) {
      const int slot = base + row;
      const float w = slotWeight[slot];
      ybuf[(size_t)slot * HID + (n0 + nfrag + ml)] = c0[r] * w;
      ybuf[(size_t)slot * HID + (n0 + nfrag + 16 + ml)] = c1[r] * w;
    }
  }
}

// ---------------------------------------------------------------------------
// 7) combine the two expert contributions per token (deterministic)
// ---------------------------------------------------------------------------
__global__ __launch_bounds__(256) void moe_combine_kernel(
    const float* __restrict__ ybuf, const int* __restrict__ tok2slot,
    float* __restrict__ out) {
  const size_t elem = ((size_t)blockIdx.x * 256 + threadIdx.x) * 4;
  const int t = (int)(elem / HID);
  const int c = (int)(elem % HID);
  const int s0 = tok2slot[t * 2 + 0];
  const int s1 = tok2slot[t * 2 + 1];
  const float4 a = *(const float4*)(ybuf + (size_t)s0 * HID + c);
  const float4 b = *(const float4*)(ybuf + (size_t)s1 * HID + c);
  float4 o;
  o.x = a.x + b.x; o.y = a.y + b.y; o.z = a.z + b.z; o.w = a.w + b.w;
  *(float4*)(out + elem) = o;
}

// ---------------------------------------------------------------------------
// host launcher
// ---------------------------------------------------------------------------
extern "C" void kernel_launch(void* const* d_in, const int* in_sizes, int n_in,
                              void* d_out, int out_size, void* d_ws,
                              size_t ws_size, hipStream_t stream) {
  const float* x      = (const float*)d_in[0];  // [2,1024,2048]
  const float* gate_w = (const float*)d_in[1];  // [8,2048]
  const float* w_gate = (const float*)d_in[2];  // [8,2048,768]
  const float* w_up   = (const float*)d_in[3];  // [8,2048,768]
  const float* w_down = (const float*)d_in[4];  // [8,768,2048]

  float* out    = (float*)d_out;                // T*H floats
  float* logits = out + (size_t)T_TOK * HID;    // T*8 floats

  char* p = (char*)d_ws;
  auto take = [&](size_t bytes) {
    char* r = p;
    p += (bytes + 255) & ~(size_t)255;
    return r;
  };
  int*    cnt        = (int*)take(NEXP * sizeof(int));
  int*    base       = (int*)take(NEXP * sizeof(int));
  int*    cursor     = (int*)take(NEXP * sizeof(int));
  int*    tokExp     = (int*)take((size_t)T_TOK * 2 * sizeof(int));
  float*  tokW       = (float*)take((size_t)T_TOK * 2 * sizeof(float));
  int*    slotToken  = (int*)take((size_t)T_TOK * 2 * sizeof(int));
  float*  slotWeight = (float*)take((size_t)T_TOK * 2 * sizeof(float));
  int*    tok2slot   = (int*)take((size_t)T_TOK * 2 * sizeof(int));
  __bf16* xbf        = (__bf16*)take((size_t)T_TOK * HID * sizeof(__bf16));
  __bf16* hbuf       = (__bf16*)take((size_t)T_TOK * 2 * INTERF * sizeof(__bf16));
  float*  ybuf       = (float*)take((size_t)T_TOK * 2 * HID * sizeof(float));
  (void)ws_size; (void)n_in; (void)in_sizes; (void)out_size;

  moe_init_kernel<<<1, 32, 0, stream>>>(cnt);
  moe_router_kernel<<<T_TOK / 8, 256, 0, stream>>>(x, gate_w, logits, tokExp,
                                                   tokW, cnt);
  moe_scan_kernel<<<1, 32, 0, stream>>>(cnt, base, cursor);
  moe_build_kernel<<<T_TOK / 256, 256, 0, stream>>>(
      tokExp, tokW, base, cursor, slotToken, slotWeight, tok2slot);
  moe_castx_kernel<<<(T_TOK * HID) / (256 * 8), 256, 0, stream>>>(x, xbf);

  dim3 g1(INTERF / BN, T_TOK / BM, NEXP);  // (12, 32, 8); early-exit past cnt[e]
  moe_mlp1_kernel<<<g1, 256, 0, stream>>>(xbf, w_gate, w_up, slotToken, base,
                                          cnt, hbuf);
  dim3 g2(HID / BN, T_TOK / BM, NEXP);     // (32, 32, 8)
  moe_mlp2_kernel<<<g2, 256, 0, stream>>>(hbuf, w_down, base, cnt, slotWeight,
                                          ybuf);
  moe_combine_kernel<<<(T_TOK * HID) / (256 * 4), 256, 0, stream>>>(
      ybuf, tok2slot, out);
}
